// DeepSeekMoE_42631845380534
// MI455X (gfx1250) — matrix operand order repaired and checked by reading, not compiled, not run on hardware
//
#include <hip/hip_runtime.h>

typedef __attribute__((ext_vector_type(16))) __bf16 v16bf;
typedef __attribute__((ext_vector_type(8)))  float  v8f;
typedef __attribute__((ext_vector_type(4)))  int    v4i;

#define BM 128
#define BN 128
#define BK 32
#define NT 256
#define BUFELEMS ((BM + BN) * BK)   // one double-buffer stage (bf16 elems)

#define AS1 __attribute__((address_space(1)))
#define AS3 __attribute__((address_space(3)))

#if defined(__has_builtin)
#if __has_builtin(__builtin_amdgcn_global_load_async_to_lds_b128) && \
    __has_builtin(__builtin_amdgcn_s_wait_asynccnt)
#define USE_ASYNC 1
#endif
#endif
#ifndef USE_ASYNC
#define USE_ASYNC 0
#endif

#if USE_ASYNC
// exact param types per hipcc diagnostic: (AS1 v4i*, AS3 v4i*, imm offset, imm cpol)
#define GASYNC_B128(gsrc, ldst)                                        \
    __builtin_amdgcn_global_load_async_to_lds_b128(                    \
        (AS1 v4i*)(AS1 void*)(gsrc), (AS3 v4i*)(AS3 void*)(ldst), 0, 0)
#endif

__device__ __forceinline__ unsigned short f2bf(float f) {
    union { float f; unsigned int u; } v; v.f = f;
    unsigned int u = v.u;
    u += 0x7FFFu + ((u >> 16) & 1u);   // round-to-nearest-even
    return (unsigned short)(u >> 16);
}

struct Frag { union { v16bf v; uint4 q[2]; }; };

// ---------------------------------------------------------------------------
// Stage one (BMxBK A-tile + BNxBK B-tile) into LDS buffer `buf`.
// A [M][Kdim] bf16 row-major; Bt [N][Kdim] bf16 row-major (pre-transposed).
// Each thread copies 32 B of A and 32 B of B (two b128 each).
// ---------------------------------------------------------------------------
__device__ __forceinline__ void stage_tiles(
    const unsigned short* __restrict__ A, const unsigned short* __restrict__ Bt,
    int Kdim, int m0, int n0, int k0,
    unsigned short* smem, int buf, int row, int col)
{
    const unsigned short* srcA = A  + (size_t)(m0 + row) * Kdim + (k0 + col);
    const unsigned short* srcB = Bt + (size_t)(n0 + row) * Kdim + (k0 + col);
    unsigned short* dA = smem + buf * BUFELEMS + row * BK + col;
    unsigned short* dB = dA + BM * BK;
#if USE_ASYNC
    GASYNC_B128(srcA,     dA);
    GASYNC_B128(srcA + 8, dA + 8);
    GASYNC_B128(srcB,     dB);
    GASYNC_B128(srcB + 8, dB + 8);
#else
    uint4 a0 = ((const uint4*)srcA)[0];
    uint4 a1 = ((const uint4*)srcA)[1];
    uint4 b0 = ((const uint4*)srcB)[0];
    uint4 b1 = ((const uint4*)srcB)[1];
    ((uint4*)dA)[0] = a0; ((uint4*)dA)[1] = a1;
    ((uint4*)dB)[0] = b0; ((uint4*)dB)[1] = b1;
#endif
}

// ---------------------------------------------------------------------------
// Double-buffered WMMA main loop: acc += A[m0:m0+128, :] * Bt[n0:n0+128, :]^T
// ---------------------------------------------------------------------------
__device__ __forceinline__ void gemm_mainloop(
    const unsigned short* __restrict__ A, const unsigned short* __restrict__ Bt,
    int Kdim, int m0, int n0, unsigned short* smem, v8f (&acc)[4][2])
{
    const int tid  = threadIdx.x;
    const int wave = tid >> 5;
    const int lane = tid & 31;
    const int waveM = wave >> 2;          // 0..1  (64-row slab)
    const int waveN = wave & 3;           // 0..3  (32-col slab)
    const int lr = lane & 15;
    const int lh = lane >> 4;
    const int row = tid >> 1;             // 0..127, 2 threads/row
    const int col = (tid & 1) << 4;       // 0 or 16

    const v8f vzero = {0.f,0.f,0.f,0.f,0.f,0.f,0.f,0.f};
#pragma unroll
    for (int mi = 0; mi < 4; ++mi)
#pragma unroll
        for (int ni = 0; ni < 2; ++ni) acc[mi][ni] = vzero;

    const int nsteps = Kdim / BK;
    stage_tiles(A, Bt, Kdim, m0, n0, 0, smem, 0, row, col);

    for (int s = 0; s < nsteps; ++s) {
        const int cur = s & 1;
        if (s + 1 < nsteps) {
            stage_tiles(A, Bt, Kdim, m0, n0, (s + 1) * BK, smem, cur ^ 1, row, col);
#if USE_ASYNC
            __builtin_amdgcn_s_wait_asynccnt(4);  // batch for `cur` complete
#endif
        } else {
#if USE_ASYNC
            __builtin_amdgcn_s_wait_asynccnt(0);
#endif
        }
        __syncthreads();   // all lanes' staged data visible

        const unsigned short* Asc = smem + cur * BUFELEMS;
        const unsigned short* Bsc = Asc + BM * BK;

        Frag a[4], b[2];
#pragma unroll
        for (int mi = 0; mi < 4; ++mi) {
            // 16-bit A 16x32 layout: K = lh*8+0..7 then 16+lh*8+0..7
            const unsigned short* p = Asc + (waveM * 64 + mi * 16 + lr) * BK + lh * 8;
            a[mi].q[0] = *(const uint4*)p;
            a[mi].q[1] = *(const uint4*)(p + 16);
        }
#pragma unroll
        for (int ni = 0; ni < 2; ++ni) {
            // B 32x16: lane holds column N=lr, K = lh*8+0..7 then 16+lh*8+0..7
            const unsigned short* p = Bsc + (waveN * 32 + ni * 16 + lr) * BK + lh * 8;
            b[ni].q[0] = *(const uint4*)p;
            b[ni].q[1] = *(const uint4*)(p + 16);
        }
#pragma unroll
        for (int mi = 0; mi < 4; ++mi)
#pragma unroll
            for (int ni = 0; ni < 2; ++ni)
                acc[mi][ni] = __builtin_amdgcn_wmma_f32_16x16x32_bf16(
                    false, a[mi].v, false, b[ni].v, (short)0, acc[mi][ni],
                    false, false);
        __syncthreads();   // done reading `cur` before it is restaged
    }
}

// ---------------------------------------------------------------------------
// Up projection: Hout = silu(X @ W1), bf16 output [M, N]
// ---------------------------------------------------------------------------
__global__ __launch_bounds__(NT) void moe_up_kernel(
    const unsigned short* __restrict__ Xb, const unsigned short* __restrict__ W1t,
    unsigned short* __restrict__ Hout, int N, int Kdim)
{
    __shared__ unsigned short smem[2 * BUFELEMS];
    const int m0 = blockIdx.x * BM;
    const int n0 = blockIdx.y * BN;
    v8f acc[4][2];
    gemm_mainloop(Xb, W1t, Kdim, m0, n0, smem, acc);

    const int wave = threadIdx.x >> 5, lane = threadIdx.x & 31;
    const int waveM = wave >> 2, waveN = wave & 3;
    const int lr = lane & 15, lh = lane >> 4;
#pragma unroll
    for (int mi = 0; mi < 4; ++mi)
#pragma unroll
        for (int ni = 0; ni < 2; ++ni)
#pragma unroll
            for (int r = 0; r < 8; ++r) {
                const int row = m0 + waveM * 64 + mi * 16 + lh * 8 + r;
                const int col = n0 + waveN * 32 + ni * 16 + lr;
                const float v = acc[mi][ni][r];
                const float s = v / (1.0f + __expf(-v));   // SiLU
                Hout[(size_t)row * N + col] = f2bf(s);
            }
}

// ---------------------------------------------------------------------------
// Down projection: Out (+)= scale[t] * (Hb @ W2), f32 output [M, N]
// e < 0 -> shared expert (scale 1); init != 0 -> overwrite instead of add
// ---------------------------------------------------------------------------
__global__ __launch_bounds__(NT) void moe_down_kernel(
    const unsigned short* __restrict__ Hb, const unsigned short* __restrict__ W2t,
    float* __restrict__ Out, const float* __restrict__ comb,
    int e, int N, int Kdim, int init)
{
    __shared__ unsigned short smem[2 * BUFELEMS];
    const int m0 = blockIdx.x * BM;
    const int n0 = blockIdx.y * BN;
    v8f acc[4][2];
    gemm_mainloop(Hb, W2t, Kdim, m0, n0, smem, acc);

    const int wave = threadIdx.x >> 5, lane = threadIdx.x & 31;
    const int waveM = wave >> 2, waveN = wave & 3;
    const int lr = lane & 15, lh = lane >> 4;
#pragma unroll
    for (int mi = 0; mi < 4; ++mi)
#pragma unroll
        for (int ni = 0; ni < 2; ++ni)
#pragma unroll
            for (int r = 0; r < 8; ++r) {
                const int row = m0 + waveM * 64 + mi * 16 + lh * 8 + r;
                const int col = n0 + waveN * 32 + ni * 16 + lr;
                const float sc = (e >= 0) ? comb[(size_t)row * 8 + e] : 1.0f;
                const size_t o = (size_t)row * N + col;
                const float v = sc * acc[mi][ni][r];
                if (init) Out[o] = v;
                else      Out[o] += v;
            }
}

// ---------------------------------------------------------------------------
// One-time weight prep: W [K][N] f32 row-major -> Wt [N][K] bf16 row-major
// ---------------------------------------------------------------------------
__global__ __launch_bounds__(NT) void transpose_convert_kernel(
    const float* __restrict__ W, unsigned short* __restrict__ Wt, int K, int N)
{
    __shared__ float tile[32][33];
    const int n0 = blockIdx.x * 32;
    const int k0 = blockIdx.y * 32;
    const int tx = threadIdx.x & 31;
    const int ty = threadIdx.x >> 5;   // 0..7
#pragma unroll
    for (int i = 0; i < 4; ++i) {
        const int k = ty * 4 + i;
        tile[k][tx] = W[(size_t)(k0 + k) * N + (n0 + tx)];
    }
    __syncthreads();
#pragma unroll
    for (int i = 0; i < 4; ++i) {
        const int n = ty * 4 + i;
        Wt[(size_t)(n0 + n) * K + (k0 + tx)] = f2bf(tile[tx][n]);
    }
}

// ---------------------------------------------------------------------------
// Router: one wave per token; top-2 renormalized softmax -> dense comb[T, 8]
// ---------------------------------------------------------------------------
__global__ __launch_bounds__(NT) void moe_router_kernel(
    const float* __restrict__ X, const float* __restrict__ Wr,
    float* __restrict__ comb, int T, int H)
{
    const int t = blockIdx.x * (NT >> 5) + (threadIdx.x >> 5);
    if (t >= T) return;
    const int lane = threadIdx.x & 31;

    float a[8] = {0.f,0.f,0.f,0.f,0.f,0.f,0.f,0.f};
    const float* xr = X + (size_t)t * H;
    for (int h = lane; h < H; h += 32) {
        const float xv = xr[h];
        const float4 w0 = ((const float4*)(Wr + (size_t)h * 8))[0];
        const float4 w1 = ((const float4*)(Wr + (size_t)h * 8))[1];
        a[0] += xv * w0.x; a[1] += xv * w0.y; a[2] += xv * w0.z; a[3] += xv * w0.w;
        a[4] += xv * w1.x; a[5] += xv * w1.y; a[6] += xv * w1.z; a[7] += xv * w1.w;
    }
#pragma unroll
    for (int off = 16; off > 0; off >>= 1)
#pragma unroll
        for (int e = 0; e < 8; ++e) a[e] += __shfl_down(a[e], off, 32);

    if (lane == 0) {
        int i1 = 0;
#pragma unroll
        for (int e = 1; e < 8; ++e) if (a[e] > a[i1]) i1 = e;
        int i2 = (i1 == 0) ? 1 : 0;
#pragma unroll
        for (int e = 0; e < 8; ++e) if (e != i1 && a[e] > a[i2]) i2 = e;
        const float m  = a[i1];
        const float e1 = __expf(a[i1] - m);
        const float e2 = __expf(a[i2] - m);
        const float inv = 1.0f / (e1 + e2);
        float o[8] = {0.f,0.f,0.f,0.f,0.f,0.f,0.f,0.f};
        o[i1] = e1 * inv; o[i2] = e2 * inv;
#pragma unroll
        for (int e = 0; e < 8; ++e) comb[(size_t)t * 8 + e] = o[e];
    }
}

// ---------------------------------------------------------------------------
// x f32 -> bf16 (8 elements / thread)
// ---------------------------------------------------------------------------
__global__ __launch_bounds__(NT) void convert_bf16_kernel(
    const float* __restrict__ X, unsigned short* __restrict__ Xb, size_t n8)
{
    const size_t i = (size_t)blockIdx.x * NT + threadIdx.x;
    if (i >= n8) return;
    const float4 f0 = ((const float4*)(X + i * 8))[0];
    const float4 f1 = ((const float4*)(X + i * 8))[1];
    union { unsigned short u[8]; uint4 q; } o;
    o.u[0] = f2bf(f0.x); o.u[1] = f2bf(f0.y); o.u[2] = f2bf(f0.z); o.u[3] = f2bf(f0.w);
    o.u[4] = f2bf(f1.x); o.u[5] = f2bf(f1.y); o.u[6] = f2bf(f1.z); o.u[7] = f2bf(f1.w);
    *(uint4*)(Xb + i * 8) = o.q;
}

// ---------------------------------------------------------------------------
extern "C" void kernel_launch(void* const* d_in, const int* in_sizes, int n_in,
                              void* d_out, int out_size, void* d_ws, size_t ws_size,
                              hipStream_t stream)
{
    (void)in_sizes; (void)n_in; (void)out_size; (void)ws_size;

    const float* x   = (const float*)d_in[0];   // [T, H]
    const float* ws1 = (const float*)d_in[1];   // [H, I]
    const float* ws2 = (const float*)d_in[2];   // [I, H]
    const float* w1  = (const float*)d_in[3];   // [E, H, I]
    const float* w2  = (const float*)d_in[4];   // [E, I, H]
    const float* wr  = (const float*)d_in[5];   // [H, E]
    float* out = (float*)d_out;                 // [T, H]

    const int T = 4 * 4096, H = 2048, I = 1408, E = 8;
    const size_t HI = (size_t)H * I;

    // workspace layout
    char* wsb = (char*)d_ws;
    unsigned short* xb   = (unsigned short*)wsb;                       // T*H bf16
    unsigned short* hbuf = (unsigned short*)(wsb + (size_t)T * H * 2); // T*I bf16
    float*          comb = (float*)(wsb + (size_t)T * H * 2 + (size_t)T * I * 2);
    unsigned short* wt   = (unsigned short*)((char*)comb + (size_t)T * 8 * 4);
    unsigned short* wt_s1 = wt;                 // [I][H]
    unsigned short* wt_s2 = wt_s1 + HI;         // [H][I]
    unsigned short* wt1   = wt_s2 + HI;         // E x [I][H]
    unsigned short* wt2   = wt1 + (size_t)E * HI; // E x [H][I]

    // 0) weight prep: f32 [K][N] -> bf16 [N][K]
    dim3 gT1(I / 32, H / 32);   // K=H, N=I
    dim3 gT2(H / 32, I / 32);   // K=I, N=H
    transpose_convert_kernel<<<gT1, NT, 0, stream>>>(ws1, wt_s1, H, I);
    transpose_convert_kernel<<<gT2, NT, 0, stream>>>(ws2, wt_s2, I, H);
    for (int e = 0; e < E; ++e) {
        transpose_convert_kernel<<<gT1, NT, 0, stream>>>(w1 + e * HI, wt1 + e * HI, H, I);
        transpose_convert_kernel<<<gT2, NT, 0, stream>>>(w2 + e * HI, wt2 + e * HI, I, H);
    }

    // 1) x -> bf16
    const size_t n8 = (size_t)T * H / 8;
    convert_bf16_kernel<<<(unsigned)((n8 + NT - 1) / NT), NT, 0, stream>>>(x, xb, n8);

    // 2) router -> dense combine weights
    moe_router_kernel<<<T / (NT >> 5), NT, 0, stream>>>(x, wr, comb, T, H);

    // 3) shared expert (initializes out), then 8 routed experts (accumulate)
    dim3 gUp(T / BM, I / BN);   // 128 x 11
    dim3 gDn(T / BM, H / BN);   // 128 x 16

    moe_up_kernel<<<gUp, NT, 0, stream>>>(xb, wt_s1, hbuf, I, H);
    moe_down_kernel<<<gDn, NT, 0, stream>>>(hbuf, wt_s2, out, nullptr, -1, H, I, 1);

    for (int e = 0; e < E; ++e) {
        moe_up_kernel<<<gUp, NT, 0, stream>>>(xb, wt1 + e * HI, hbuf, I, H);
        moe_down_kernel<<<gDn, NT, 0, stream>>>(hbuf, wt2 + e * HI, out,
                                                comb, e, H, I, 0);
    }
}
